// DeepseekV3Attention_15324443312731
// MI455X (gfx1250) — compile-verified, matrix operand order checked by reading, not run on hardware
//
#include <hip/hip_runtime.h>

// ---------------- problem constants (fixed by setup_inputs) ----------------
#define BSZ      16
#define QLORA    1536
#define KVLORA   512
#define CAP      4096
#define STARTPOS 2048
#define KVLEN    2049          // STARTPOS + q_len
#define HEADS    128
#define DQ       192           // per-head q dim (128 nope + 64 rope)
#define DNOPE    128
#define DROPE    64
#define VHEAD    128
#define QCOLS    (HEADS * DQ)  // 24576
#define SSTRIDE  2064          // scores row stride (2049 padded to mult of 16)
#define SCALE    0.07216878364870323f   // 1/sqrt(192)

typedef __attribute__((ext_vector_type(16))) __bf16 v16bf;
typedef __attribute__((ext_vector_type(8)))  float  v8f;

// ---------------- helpers ----------------
__device__ __forceinline__ __bf16 to_bf16(float f) { return (__bf16)f; }

__device__ __forceinline__ int kmap(int e, int kh) {
  // bf16 16x32 A/B fragment K index for element e in lane-half kh (ISA 7.12.2)
  return (e < 8 ? e : e + 8) + (kh << 3);
}

__device__ __forceinline__ v8f wmma_bf16(v16bf a, v16bf b, v8f c) {
  return __builtin_amdgcn_wmma_f32_16x16x32_bf16(false, a, false, b,
                                                 (short)0, c, false, false);
}

// ---- CDNA5 async copy: global -> LDS, 16B per lane, ASYNCcnt tracked ------
#define ASYNC_B128(loff, gptr)                                               \
  asm volatile("global_load_async_to_lds_b128 %0, %1, off"                   \
               :: "v"(loff), "v"(gptr) : "memory")
#define WAIT_ASYNC_4() asm volatile("s_wait_asynccnt 4" ::: "memory")
#define WAIT_ASYNC_0() asm volatile("s_wait_asynccnt 0" ::: "memory")

__device__ __forceinline__ unsigned lds_off(const float* p) {
  return (unsigned)(size_t)p;    // generic LDS address truncates to LDS offset
}

// stage 16 rows x 32 f32 cols from row-major [R][stride] into LDS tile [16][32]
__device__ __forceinline__ void stage_rows(unsigned loff,
                                           const float* __restrict__ base,
                                           int stride, int k0, int lane) {
  int seg = lane & 7, rb = lane >> 3;
#pragma unroll
  for (int i = 0; i < 4; ++i) {
    int r = i * 4 + rb;
    const float* g = base + (size_t)r * stride + k0 + seg * 4;
    ASYNC_B128(loff + (unsigned)(r * 128 + seg * 16), g);
  }
}

// same, but rows are kv-cache positions with the n==STARTPOS row redirected
__device__ __forceinline__ void stage_kv_rows(unsigned loff,
                                              const float* __restrict__ cache,
                                              const float* __restrict__ newrow,
                                              int rowlen, int n0, int k0, int lane) {
  int seg = lane & 7, rb = lane >> 3;
#pragma unroll
  for (int i = 0; i < 4; ++i) {
    int r = i * 4 + rb;
    int n = n0 + r;                                   // n < CAP always (valid mem)
    const float* row = (n == STARTPOS) ? newrow : cache + (size_t)n * rowlen;
    ASYNC_B128(loff + (unsigned)(r * 128 + seg * 16), row + k0 + seg * 4);
  }
}

// stage 32 k-rows x 16 f32 cols from row-major [K][stride] into LDS tile [32][16]
__device__ __forceinline__ void stage_cols(unsigned loff,
                                           const float* __restrict__ base,
                                           int stride, int k0, int n0, int lane) {
#pragma unroll
  for (int i = 0; i < 4; ++i) {
    const float* g = base + (size_t)(k0 + lane) * stride + n0 + i * 4;
    ASYNC_B128(loff + (unsigned)(lane * 64 + i * 16), g);
  }
}

// ---------------- fragment builders ----------------
// A-style fragment from row-major [16][K] global memory (also B from [N][K])
__device__ __forceinline__ v16bf frag_nk(const float* __restrict__ base,
                                         int stride, int k0, int lane) {
  int r = lane & 15, kh = (lane >> 4) & 1;
  const float* p = base + (size_t)r * stride + k0;
  v16bf out;
#pragma unroll
  for (int e = 0; e < 16; ++e) out[e] = to_bf16(p[kmap(e, kh)]);
  return out;
}

// B fragment from LDS tile [16 cols][32 k] (pitch 32): lane's B-column = tile row
__device__ __forceinline__ v16bf frag_ldsT(const float* __restrict__ t, int lane) {
  const float* p = t + (lane & 15) * 32;
  int kh = (lane >> 4) & 1;
  v16bf out;
#pragma unroll
  for (int e = 0; e < 16; ++e) out[e] = to_bf16(p[kmap(e, kh)]);
  return out;
}

// B fragment from LDS tile [32 k][16 cols] (pitch 16)
__device__ __forceinline__ v16bf frag_ldsK(const float* __restrict__ t, int lane) {
  int n = lane & 15, kh = (lane >> 4) & 1;
  v16bf out;
#pragma unroll
  for (int e = 0; e < 16; ++e) out[e] = to_bf16(t[kmap(e, kh) * 16 + n]);
  return out;
}

// ---------------- K1: Q projection  q[b][n] = sum_l qn[b,l] * wq_b[n,l] ----
__global__ void k_qproj(const float* __restrict__ qn,
                        const float* __restrict__ wqb,
                        float* __restrict__ ws_q) {
  __shared__ float lds[8192];                      // 8 waves * 2 bufs * 512 f32
  int lane = threadIdx.x & 31, wave = threadIdx.x >> 5;
  float* buf = &lds[wave * 1024];
  unsigned l0 = lds_off(&buf[0]), l1 = lds_off(&buf[512]);
  int n0 = (blockIdx.x * 8 + wave) * 16;           // 192 blocks * 8 waves = 1536 tiles
  const float* B = wqb + (size_t)n0 * QLORA;
  v8f acc = {};
  const int NS = QLORA / 32;                       // 48
  stage_rows(l0, B, QLORA, 0, lane);
  for (int s = 0; s < NS; ++s) {
    if (s + 1 < NS) { stage_rows((s & 1) ? l0 : l1, B, QLORA, (s + 1) * 32, lane);
                      WAIT_ASYNC_4(); }
    else            { WAIT_ASYNC_0(); }
    v16bf a  = frag_nk(qn, QLORA, s * 32, lane);   // A: [16 batch][K]
    v16bf bb = frag_ldsT((s & 1) ? &buf[512] : &buf[0], lane);
    acc = wmma_bf16(a, bb, acc);
  }
  int col = lane & 15, kh = lane >> 4;
#pragma unroll
  for (int v = 0; v < 8; ++v)
    ws_q[(size_t)(v + 8 * kh) * QCOLS + n0 + col] = acc[v];
}

// ---------------- K2: RoPE (interleaved) on q_pe and k_pe ------------------
__global__ void k_rope(const float* __restrict__ kpe_in,
                       const float* __restrict__ sinc,
                       const float* __restrict__ cosc,
                       float* __restrict__ ws_q,
                       float* __restrict__ ws_kpe) {
  int h = blockIdx.x, b = blockIdx.y, j = threadIdx.x;   // 32 threads
  float cs = cosc[(size_t)STARTPOS * DROPE + j];
  float sn = sinc[(size_t)STARTPOS * DROPE + j];
  float* qpe = ws_q + (size_t)b * QCOLS + h * DQ + DNOPE;
  float x0 = qpe[2 * j], x1 = qpe[2 * j + 1];
  __syncthreads();
  qpe[j]      = x0 * cs - x1 * sn;
  qpe[32 + j] = x1 * cs + x0 * sn;
  if (h == 0) {
    const float* kp = kpe_in + (size_t)b * DROPE;
    float y0 = kp[2 * j], y1 = kp[2 * j + 1];
    ws_kpe[b * DROPE + j]      = y0 * cs - y1 * sn;
    ws_kpe[b * DROPE + 32 + j] = y1 * cs + y0 * sn;
  }
}

// ---------------- K3: absorb  q_abs[b,h,c] = sum_d q_nope[b,h,d]*wkv_b[h,d,c]
__global__ void k_absorb(const float* __restrict__ ws_q,
                         const float* __restrict__ wkv_b,
                         float* __restrict__ ws_qabs) {
  __shared__ float lds[8192];
  int lane = threadIdx.x & 31, wave = threadIdx.x >> 5;
  float* buf = &lds[wave * 1024];
  unsigned l0 = lds_off(&buf[0]), l1 = lds_off(&buf[512]);
  int task = blockIdx.x * 8 + wave;                // 512 blocks -> 4096 = 128h*32ct
  int h = task >> 5, n0 = (task & 31) * 16;
  const float* A  = ws_q + (size_t)h * DQ;         // [16 batch][K=128], stride QCOLS
  const float* Bp = wkv_b + (size_t)h * 256 * KVLORA;   // [K=d][N=c] row-major
  v8f acc = {};
  const int NS = DNOPE / 32;                       // 4
  stage_cols(l0, Bp, KVLORA, 0, n0, lane);
  for (int s = 0; s < NS; ++s) {
    if (s + 1 < NS) { stage_cols((s & 1) ? l0 : l1, Bp, KVLORA, (s + 1) * 32, n0, lane);
                      WAIT_ASYNC_4(); }
    else            { WAIT_ASYNC_0(); }
    v16bf a  = frag_nk(A, QCOLS, s * 32, lane);
    v16bf bb = frag_ldsK((s & 1) ? &buf[512] : &buf[0], lane);
    acc = wmma_bf16(a, bb, acc);
  }
  int col = lane & 15, kh = lane >> 4;
#pragma unroll
  for (int v = 0; v < 8; ++v)
    ws_qabs[(size_t)(v + 8 * kh) * (HEADS * KVLORA) + h * KVLORA + n0 + col] = acc[v];
}

// ---------------- K4: scores = (q_abs . ckv + q_pe . kpe) * SCALE ----------
__global__ void k_scores(const float* __restrict__ ws_qabs,
                         const float* __restrict__ ws_q,
                         const float* __restrict__ ckv_cache,
                         const float* __restrict__ ckv_new,
                         const float* __restrict__ kpe_cache,
                         const float* __restrict__ kpe_new,
                         float* __restrict__ ws_scores) {
  __shared__ float lds[8192];
  int lane = threadIdx.x & 31, wave = threadIdx.x >> 5;
  float* buf = &lds[wave * 1024];
  unsigned l0 = lds_off(&buf[0]), l1 = lds_off(&buf[512]);
  int h0 = blockIdx.x * 16, b = blockIdx.y;
  const float* A1 = ws_qabs + (size_t)b * HEADS * KVLORA + h0 * KVLORA; // stride 512
  const float* A2 = ws_q + (size_t)b * QCOLS + h0 * DQ + DNOPE;        // stride 192
  const float* kvb = ckv_cache + (size_t)b * CAP * KVLORA;
  const float* kpb = kpe_cache + (size_t)b * CAP * DROPE;
  const float* kvn = ckv_new + (size_t)b * KVLORA;
  const float* kpn = kpe_new + (size_t)b * DROPE;
  const int NS = 18;                               // 16 nope + 2 rope k-steps
  for (int nt = wave; nt < 129; nt += 8) {
    int n0 = nt * 16;
    v8f acc = {};
    stage_kv_rows(l0, kvb, kvn, KVLORA, n0, 0, lane);
    for (int s = 0; s < NS; ++s) {
      if (s + 1 < NS) {
        unsigned lt = (s & 1) ? l0 : l1;
        int sn = s + 1;
        if (sn < 16) stage_kv_rows(lt, kvb, kvn, KVLORA, n0, sn * 32, lane);
        else         stage_kv_rows(lt, kpb, kpn, DROPE,  n0, (sn - 16) * 32, lane);
        WAIT_ASYNC_4();
      } else { WAIT_ASYNC_0(); }
      v16bf a = (s < 16) ? frag_nk(A1, KVLORA, s * 32, lane)
                         : frag_nk(A2, DQ, (s - 16) * 32, lane);
      v16bf bb = frag_ldsT((s & 1) ? &buf[512] : &buf[0], lane);
      acc = wmma_bf16(a, bb, acc);
    }
    int col = lane & 15, kh = lane >> 4;
    int n = n0 + col;
    if (n < KVLEN) {
#pragma unroll
      for (int v = 0; v < 8; ++v)
        ws_scores[((size_t)b * HEADS + h0 + v + 8 * kh) * SSTRIDE + n] = acc[v] * SCALE;
    }
  }
}

// ---------------- K5: row softmax over kv_len=2049 -------------------------
__global__ void k_softmax(float* __restrict__ ws_scores) {
  float* s = ws_scores + (size_t)blockIdx.x * SSTRIDE;
  int t = threadIdx.x;
  __shared__ float red[8];
  float m = -1e30f;
  for (int i = t; i < KVLEN; i += 256) m = fmaxf(m, s[i]);
  for (int o = 16; o > 0; o >>= 1) m = fmaxf(m, __shfl_xor(m, o, 32));
  if ((t & 31) == 0) red[t >> 5] = m;
  __syncthreads();
  float mx = red[0];
#pragma unroll
  for (int w = 1; w < 8; ++w) mx = fmaxf(mx, red[w]);
  __syncthreads();
  float sum = 0.f;
  for (int i = t; i < KVLEN; i += 256) { float e = __expf(s[i] - mx); s[i] = e; sum += e; }
  for (int o = 16; o > 0; o >>= 1) sum += __shfl_xor(sum, o, 32);
  if ((t & 31) == 0) red[t >> 5] = sum;
  __syncthreads();
  float tot = 0.f;
#pragma unroll
  for (int w = 0; w < 8; ++w) tot += red[w];
  float inv = 1.0f / tot;
  for (int i = t; i < KVLEN; i += 256) s[i] *= inv;
  for (int i = KVLEN + t; i < SSTRIDE; i += 256) s[i] = 0.0f;   // clean padding
}

// ---------------- K6: o[b,h,c] = sum_n probs[b,h,n] * kv[n,c] --------------
__global__ void k_ogemm(const float* __restrict__ ws_scores,
                        const float* __restrict__ ckv_cache,
                        const float* __restrict__ ckv_new,
                        float* __restrict__ ws_o) {
  __shared__ float lds[8192];
  int lane = threadIdx.x & 31, wave = threadIdx.x >> 5;
  float* buf = &lds[wave * 1024];
  unsigned l0 = lds_off(&buf[0]), l1 = lds_off(&buf[512]);
  int b = blockIdx.z, h0 = blockIdx.y * 16;
  int c0 = (blockIdx.x * 8 + wave) * 16;           // 4*8 = 32 c-tiles
  const float* A   = ws_scores + ((size_t)b * HEADS + h0) * SSTRIDE;
  const float* kvb = ckv_cache + (size_t)b * CAP * KVLORA;
  const float* kvn = ckv_new + (size_t)b * KVLORA;
  v8f acc = {};
  const int NS = STARTPOS / 32;                    // 64 steps: n = 0..2047
  stage_cols(l0, kvb, KVLORA, 0, c0, lane);
  for (int s = 0; s < NS; ++s) {
    if (s + 1 < NS) { stage_cols((s & 1) ? l0 : l1, kvb, KVLORA, (s + 1) * 32, c0, lane);
                      WAIT_ASYNC_4(); }
    else            { WAIT_ASYNC_0(); }
    v16bf a  = frag_nk(A, SSTRIDE, s * 32, lane);
    v16bf bb = frag_ldsK((s & 1) ? &buf[512] : &buf[0], lane);
    acc = wmma_bf16(a, bb, acc);
  }
  int col = lane & 15, kh = lane >> 4;
  float kvv = kvn[c0 + col];                       // n = 2048 : fresh row
#pragma unroll
  for (int v = 0; v < 8; ++v)
    acc[v] += A[(size_t)(v + 8 * kh) * SSTRIDE + STARTPOS] * kvv;
#pragma unroll
  for (int v = 0; v < 8; ++v)
    ws_o[(size_t)b * HEADS * KVLORA + (h0 + v + 8 * kh) * KVLORA + c0 + col] = acc[v];
}

// ---------------- K7: out[b,h,d] = sum_c o[b,h,c] * wkv_b[h,128+d,c] -------
__global__ void k_outproj(const float* __restrict__ ws_o,
                          const float* __restrict__ wkv_b,
                          float* __restrict__ out) {
  __shared__ float lds[8192];
  int lane = threadIdx.x & 31, wave = threadIdx.x >> 5;
  float* buf = &lds[wave * 1024];
  unsigned l0 = lds_off(&buf[0]), l1 = lds_off(&buf[512]);
  int task = blockIdx.x * 8 + wave;                // 128 blocks -> 1024 = 128h*8dt
  int h = task >> 3, n0 = (task & 7) * 16;
  const float* A  = ws_o + (size_t)h * KVLORA;     // [16 batch][K=512], stride H*KVLORA
  const float* Bp = wkv_b + (size_t)h * 256 * KVLORA + (size_t)(DNOPE + n0) * KVLORA;
  v8f acc = {};
  const int NS = KVLORA / 32;                      // 16
  stage_rows(l0, Bp, KVLORA, 0, lane);
  for (int s = 0; s < NS; ++s) {
    if (s + 1 < NS) { stage_rows((s & 1) ? l0 : l1, Bp, KVLORA, (s + 1) * 32, lane);
                      WAIT_ASYNC_4(); }
    else            { WAIT_ASYNC_0(); }
    v16bf a  = frag_nk(A, HEADS * KVLORA, s * 32, lane);
    v16bf bb = frag_ldsT((s & 1) ? &buf[512] : &buf[0], lane);
    acc = wmma_bf16(a, bb, acc);
  }
  int col = lane & 15, kh = lane >> 4;
#pragma unroll
  for (int v = 0; v < 8; ++v)
    out[(size_t)(v + 8 * kh) * (HEADS * VHEAD) + h * VHEAD + n0 + col] = acc[v];
}

// ---------------- launch ---------------------------------------------------
extern "C" void kernel_launch(void* const* d_in, const int* in_sizes, int n_in,
                              void* d_out, int out_size, void* d_ws, size_t ws_size,
                              hipStream_t stream) {
  const float* qn        = (const float*)d_in[0];   // (16,1,1536)
  const float* ckv_new   = (const float*)d_in[1];   // (16,1,512)  == new cache row
  const float* kpe_in    = (const float*)d_in[2];   // (16,1,1,64)
  const float* ckv_cache = (const float*)d_in[5];   // (16,4096,512)  (not mutated)
  const float* kpe_cache = (const float*)d_in[6];   // (16,4096,64)
  const float* sinc      = (const float*)d_in[7];   // (4096,64)
  const float* cosc      = (const float*)d_in[8];   // (4096,64)
  const float* wkv_b     = (const float*)d_in[9];   // (128,256,512)
  const float* wq_b      = (const float*)d_in[10];  // (24576,1536)
  float* out = (float*)d_out;

  float* ws        = (float*)d_ws;
  float* ws_q      = ws;                            // 16*24576     = 393216
  float* ws_kpe    = ws + 393216;                   // 16*64        = 1024
  float* ws_qabs   = ws + 394240;                   // 16*128*512   = 1048576
  float* ws_scores = ws + 1442816;                  // 16*128*2064  = 4227072
  float* ws_o      = ws + 5669888;                  // 16*128*512   = 1048576

  k_qproj  <<<dim3(192),         dim3(256), 0, stream>>>(qn, wq_b, ws_q);
  k_rope   <<<dim3(HEADS, BSZ),  dim3(32),  0, stream>>>(kpe_in, sinc, cosc, ws_q, ws_kpe);
  k_absorb <<<dim3(512),         dim3(256), 0, stream>>>(ws_q, wkv_b, ws_qabs);
  k_scores <<<dim3(8, BSZ),      dim3(256), 0, stream>>>(ws_qabs, ws_q, ckv_cache, ckv_new,
                                                         kpe_cache, ws_kpe, ws_scores);
  k_softmax<<<dim3(BSZ * HEADS), dim3(256), 0, stream>>>(ws_scores);
  k_ogemm  <<<dim3(4, 8, BSZ),   dim3(256), 0, stream>>>(ws_scores, ckv_cache, ckv_new, ws_o);
  k_outproj<<<dim3(128),         dim3(256), 0, stream>>>(ws_o, wkv_b, out);
}
